// AdaptiveSegmenter_24739011625045
// MI455X (gfx1250) — matrix operand order repaired and checked by reading, not compiled, not run on hardware
//
#include <hip/hip_runtime.h>
#include <hip/hip_bf16.h>
#include <math.h>
#include <stdint.h>

#define Bb 8
#define Tt 4096
#define Dd 512
#define G4 2048   // 4*D
#define H1 256    // D/2

// ---- workspace layout (bytes), all 16B aligned ----
static const size_t WS_GATES = 0;                        // 8*4096*2048*4 = 268435456
static const size_t WS_BND   = 268435456;                // 8*4096
static const size_t WS_DEST  = WS_BND + 32768;           // 8*4096*4
static const size_t WS_FB    = WS_DEST + 131072;         // 8*4
static const size_t WS_MEAN  = WS_FB + 32;               // 8*512*4
static const size_t WS_WHBF  = WS_MEAN + 16384;          // 2048*512*2
static const size_t WS_XBF   = WS_WHBF + 2097152;        // 32768*512*2
static const size_t WS_W1BF  = WS_XBF + 33554432;        // 256*512*2
static const size_t WS_WIHBF = WS_W1BF + 262144;         // 2048*512*2

typedef __attribute__((ext_vector_type(16))) __bf16 v16bf;
typedef __attribute__((ext_vector_type(8)))  float  v8f;

union Frag { v16bf v; unsigned int u[8]; };

__device__ __forceinline__ unsigned short f2bf(float f) {
  unsigned int u = __float_as_uint(f);
  u += 0x7FFFu + ((u >> 16) & 1u);       // round-to-nearest-even
  return (unsigned short)(u >> 16);
}
__device__ __forceinline__ unsigned int pk2bf(float lo, float hi) {
  return (unsigned int)f2bf(lo) | ((unsigned int)f2bf(hi) << 16);
}
// A fragment (16x32 bf16): u[0..3] = K[half*8 .. +7], u[4..7] = K[16+half*8 .. +7]
__device__ __forceinline__ void loadA(Frag& f, const unsigned short* rowk0, int half) {
  uint4 lo = *(const uint4*)(rowk0 + half * 8);
  uint4 hi = *(const uint4*)(rowk0 + 16 + half * 8);
  f.u[0] = lo.x; f.u[1] = lo.y; f.u[2] = lo.z; f.u[3] = lo.w;
  f.u[4] = hi.x; f.u[5] = hi.y; f.u[6] = hi.z; f.u[7] = hi.w;
}
// B fragment (32x16 bf16): u[0..3] = K[half*8 .. +7], u[4..7] = K[16+half*8 .. +7]
__device__ __forceinline__ void loadB(Frag& f, const unsigned short* colk0, int half) {
  uint4 lo = *(const uint4*)(colk0 + half * 8);
  uint4 hi = *(const uint4*)(colk0 + 16 + half * 8);
  f.u[0] = lo.x; f.u[1] = lo.y; f.u[2] = lo.z; f.u[3] = lo.w;
  f.u[4] = hi.x; f.u[5] = hi.y; f.u[6] = hi.z; f.u[7] = hi.w;
}
__device__ __forceinline__ float sigf(float x) { return 1.0f / (1.0f + __expf(-x)); }

// CDNA5 async global->LDS copy (ASYNCcnt-tracked), 16 bytes per lane
__device__ __forceinline__ void async_ld_b128(unsigned int lds_addr, const void* gaddr) {
  asm volatile("global_load_async_to_lds_b128 %0, %1, off"
               :: "v"(lds_addr), "v"((unsigned long long)(uintptr_t)gaddr)
               : "memory");
}
template <int N>
__device__ __forceinline__ void wait_asynccnt() {
  asm volatile("s_wait_asynccnt %0" :: "i"(N) : "memory");
}

// ---------- K0: f32 -> packed bf16 converter ----------
__global__ void k_cvt_bf16(const float* __restrict__ in, unsigned int* __restrict__ out,
                           int npairs) {
  int i = blockIdx.x * blockDim.x + threadIdx.x;
  if (i < npairs) {
    float2 f = *(const float2*)(in + 2 * (size_t)i);
    out[i] = pk2bf(f.x, f.y);
  }
}

// ---------- K1: zero padded output ----------
__global__ void k_zero(float4* __restrict__ p, int n4) {
  int stride = gridDim.x * blockDim.x;
  for (int i = blockIdx.x * blockDim.x + threadIdx.x; i < n4; i += stride)
    p[i] = make_float4(0.f, 0.f, 0.f, 0.f);
}

// ---------- K2: boundary MLP (WMMA bf16) ----------
__global__ __launch_bounds__(128) void k_boundary_mlp(
    const unsigned short* __restrict__ xbf, const unsigned short* __restrict__ w1bf,
    const float* __restrict__ b1, const float* __restrict__ w2,
    const float* __restrict__ b2, float* __restrict__ logits_out,
    unsigned char* __restrict__ bnd) {
  __shared__ float h1s[16][H1 + 1];
  __shared__ float sl[2][16];
  int lane = threadIdx.x & 31, wv = threadIdx.x >> 5;
  int half = lane >> 4, ln = lane & 15;
  int m0 = blockIdx.x * 16;       // 16 tokens per workgroup
  int n0w = wv * 64;              // each of 4 waves covers 64 h1 cols
  v8f acc[4];
  for (int i = 0; i < 4; i++) acc[i] = (v8f)0.0f;
  for (int ks = 0; ks < 16; ks++) {
    int k0 = ks * 32;
    Frag a, b[4];
    loadA(a, xbf + (size_t)(m0 + ln) * Dd + k0, half);
    for (int nt = 0; nt < 4; nt++)
      loadB(b[nt], w1bf + (size_t)(n0w + nt * 16 + ln) * Dd + k0, half);
    for (int nt = 0; nt < 4; nt++)
      acc[nt] = __builtin_amdgcn_wmma_f32_16x16x32_bf16(
          false, a.v, false, b[nt].v, (short)0, acc[nt], false, false);
  }
  for (int nt = 0; nt < 4; nt++) {
    int n = n0w + nt * 16 + ln;
    float bias = b1[n];
    for (int v = 0; v < 8; v++) {
      float val = acc[nt][v] + bias;
      val = 0.5f * val * (1.0f + erff(val * 0.70710678118654752f));  // exact GELU
      h1s[half * 8 + v][n] = val;
    }
  }
  __syncthreads();
  int tid = threadIdx.x;
  if (tid < 32) {
    int tok = tid & 15, g = tid >> 4;
    float s = b2[g];
    const float* wr = w2 + g * H1;
    for (int h = 0; h < H1; h++) s += h1s[tok][h] * wr[h];
    int row = m0 + tok;
    logits_out[(size_t)row * 2 + g] = s;
    sl[g][tok] = s;
  }
  __syncthreads();
  if (tid < 16) {
    int row = m0 + tid;
    int t = row & (Tt - 1);
    bool forced = (((t & 31) == 0) && (t != 0)) || (t == Tt - 1);
    bnd[row] = (forced || (sl[1][tid] > sl[0][tid])) ? 1 : 0;
  }
}

// ---------- K3: gates_x GEMM with async-to-LDS double-buffered pipeline ----------
__global__ __launch_bounds__(256) void k_gates_gemm(
    const unsigned short* __restrict__ xbf, const unsigned short* __restrict__ wihbf,
    const float* __restrict__ bih, const float* __restrict__ bhh,
    float* __restrict__ gates) {
  __shared__ __align__(16) unsigned short Asl[2][128][40];  // 128 rows x 32 k (+8 pad)
  __shared__ __align__(16) unsigned short Bsl[2][256][40];  // 256 cols x 32 k (+8 pad)
  int tid = threadIdx.x;
  int lane = tid & 31, wv = tid >> 5;
  int half = lane >> 4, ln = lane & 15;
  int mblk = blockIdx.x * 128, nblk = blockIdx.y * 256;
  int mw = (wv & 1) * 64, nw = (wv >> 1) * 64;

  auto preload = [&](int buf, int ks) {
    int k0 = ks * 32;
    #pragma unroll
    for (int r = 0; r < 2; r++) {                // A tile: 512 x 16B chunks
      int c = tid + r * 256;
      int row = c >> 2, seg = c & 3;
      async_ld_b128((unsigned int)(uintptr_t)&Asl[buf][row][seg * 8],
                    xbf + (size_t)(mblk + row) * Dd + k0 + seg * 8);
    }
    #pragma unroll
    for (int r = 0; r < 4; r++) {                // B tile: 1024 x 16B chunks
      int c = tid + r * 256;
      int row = c >> 2, seg = c & 3;
      async_ld_b128((unsigned int)(uintptr_t)&Bsl[buf][row][seg * 8],
                    wihbf + (size_t)(nblk + row) * Dd + k0 + seg * 8);
    }
  };  // 6 async b128 per thread per buffer

  v8f acc[16];
  for (int i = 0; i < 16; i++) acc[i] = (v8f)0.0f;
  preload(0, 0);
  for (int ks = 0; ks < 16; ks++) {
    int buf = ks & 1;
    if (ks < 15) { preload(buf ^ 1, ks + 1); wait_asynccnt<6>(); }
    else         { wait_asynccnt<0>(); }
    __syncthreads();                             // current buffer visible to all waves
    Frag a[4], b[4];
    for (int mt = 0; mt < 4; mt++) loadA(a[mt], &Asl[buf][mw + mt * 16 + ln][0], half);
    for (int nt = 0; nt < 4; nt++) loadB(b[nt], &Bsl[buf][nw + nt * 16 + ln][0], half);
    for (int mt = 0; mt < 4; mt++)
      for (int nt = 0; nt < 4; nt++)
        acc[mt * 4 + nt] = __builtin_amdgcn_wmma_f32_16x16x32_bf16(
            false, a[mt].v, false, b[nt].v, (short)0, acc[mt * 4 + nt], false, false);
    __syncthreads();                             // done reading before next overwrite
  }
  for (int nt = 0; nt < 4; nt++) {
    int n = nblk + nw + nt * 16 + ln;
    float bias = bih[n] + bhh[n];
    for (int mt = 0; mt < 4; mt++) {
      int mbase = mblk + mw + mt * 16 + half * 8;
      v8f d = acc[mt * 4 + nt];
      for (int v = 0; v < 8; v++)
        gates[(size_t)(mbase + v) * G4 + n] = d[v] + bias;
    }
  }
}

// ---------- K4: boundary scan -> dest, bidx, fallback flags ----------
__global__ void k_scan(const unsigned char* __restrict__ bnd,
                       int* __restrict__ dest, int* __restrict__ fb,
                       float* __restrict__ bidx_out) {
  int b = threadIdx.x;
  if (b >= Bb) return;
  const unsigned char* br = bnd + (size_t)b * Tt;
  int* dr = dest + (size_t)b * Tt;
  float* bo = bidx_out + (size_t)b * Tt;
  int j = 0;
  unsigned char prev = 0;
  for (int t = 0; t < Tt; t++) {
    unsigned char isb = br[t];
    bool reset = (t == 0) || (prev != 0);
    bool emit = (isb != 0) && !reset;
    if (emit) { dr[t] = j; bo[j] = (float)t; j++; }
    else dr[t] = -1;
    prev = isb;
  }
  for (int r = j; r < Tt; r++) bo[r] = -1.0f;
  if (j == 0) { bo[0] = (float)(Tt - 1); fb[b] = 1; } else fb[b] = 0;
}

// ---------- K5: mean over time (fallback source) ----------
__global__ void k_mean(const float* __restrict__ x, float* __restrict__ mean) {
  int b = blockIdx.x, d = threadIdx.x;
  const float* xp = x + (size_t)b * Tt * Dd + d;
  float s = 0.f;
  for (int t = 0; t < Tt; t++) s += xp[(size_t)t * Dd];
  mean[b * Dd + d] = s * (1.0f / Tt);
}

// ---------- K6: fallback fill ----------
__global__ void k_fallback(const int* __restrict__ fb, const float* __restrict__ mean,
                           float* __restrict__ padded) {
  int b = blockIdx.x, d = threadIdx.x;
  if (fb[b]) padded[(size_t)b * Tt * Dd + d] = mean[b * Dd + d];
}

// ---------- K7: segmented LSTM recurrence, 1 WG per 32-step region ----------
__global__ __launch_bounds__(512) void k_seg_lstm(
    const unsigned short* __restrict__ whbf,   // W_hh bf16 [2048][512]
    const float* __restrict__ gates,           // gates_x [B][T][2048]
    const unsigned char* __restrict__ bnd,
    const int* __restrict__ dest,
    float* __restrict__ padded) {
  __shared__ unsigned short hbf[16][520];      // h (bf16), [batch(+pad)][k]; 1040B rows
  int tid = threadIdx.x;
  int lane = tid & 31, wv = tid >> 5;          // 16 waves
  int half = lane >> 4, n = lane & 15;         // n = batch column (0..7 real)
  int uw = wv * 32;                            // wave owns hidden units [uw, uw+32)
  for (int i = tid; i < 16 * 520; i += 512) ((unsigned short*)hbf)[i] = 0;
  float creg[16];
  for (int i = 0; i < 16; i++) creg[i] = 0.f;
  int rk = blockIdx.x;
  int s = (rk == 0) ? 0 : (32 * rk + 1);       // region start (guaranteed reset)
  int e = 32 * (rk + 1); if (e > Tt - 1) e = Tt - 1;
  for (int t = s; t <= e; t++) {
    __syncthreads();                           // h in LDS valid for this step
    // ---- issue this step's gate loads up front; they overlap the WMMA chain ----
    float4 gv[2][8];
    bool resetv = true, bndc = false; int dst = -1;
    if (n < Bb) {
      resetv = (t == s) ? true : (bnd[(size_t)n * Tt + t - 1] != 0);
      bndc = bnd[(size_t)n * Tt + t] != 0;     // == reset[t+1]
      dst = dest[(size_t)n * Tt + t];
      const float* gx = gates + ((size_t)n * Tt + t) * (size_t)G4;
      #pragma unroll
      for (int r = 0; r < 2; r++) {
        const float* g0 = gx + uw + r * 16 + half * 8;
        gv[r][0] = *(const float4*)(g0);          gv[r][1] = *(const float4*)(g0 + 4);
        gv[r][2] = *(const float4*)(g0 + 512);    gv[r][3] = *(const float4*)(g0 + 516);
        gv[r][4] = *(const float4*)(g0 + 1024);   gv[r][5] = *(const float4*)(g0 + 1028);
        gv[r][6] = *(const float4*)(g0 + 1536);   gv[r][7] = *(const float4*)(g0 + 1540);
      }
    }
    // ---- W_hh @ h via WMMA, K=512 ----
    v8f acc[8];                                // [gate q][row-tile r]
    for (int i = 0; i < 8; i++) acc[i] = (v8f)0.0f;
    for (int ks = 0; ks < 16; ks++) {
      int k0 = ks * 32;
      Frag bf;
      loadB(bf, &hbf[n][k0], half);            // 2x ds_load_b128
      for (int q = 0; q < 4; q++)
        for (int r = 0; r < 2; r++) {
          Frag af;
          loadA(af, whbf + (size_t)(q * Dd + uw + r * 16 + n) * Dd + k0, half);
          acc[q * 2 + r] = __builtin_amdgcn_wmma_f32_16x16x32_bf16(
              false, af.v, false, bf.v, (short)0, acc[q * 2 + r], false, false);
        }
    }
    unsigned int hpack[8];                     // bf16 h to write back to LDS
    if (n < Bb) {
      #pragma unroll
      for (int r = 0; r < 2; r++) {
        int ubase = uw + r * 16 + half * 8;    // 8 consecutive units
        float gi[8] = {gv[r][0].x, gv[r][0].y, gv[r][0].z, gv[r][0].w,
                       gv[r][1].x, gv[r][1].y, gv[r][1].z, gv[r][1].w};
        float gf[8] = {gv[r][2].x, gv[r][2].y, gv[r][2].z, gv[r][2].w,
                       gv[r][3].x, gv[r][3].y, gv[r][3].z, gv[r][3].w};
        float gg[8] = {gv[r][4].x, gv[r][4].y, gv[r][4].z, gv[r][4].w,
                       gv[r][5].x, gv[r][5].y, gv[r][5].z, gv[r][5].w};
        float go[8] = {gv[r][6].x, gv[r][6].y, gv[r][6].z, gv[r][6].w,
                       gv[r][7].x, gv[r][7].y, gv[r][7].z, gv[r][7].w};
        float hn[8];
        #pragma unroll
        for (int v = 0; v < 8; v++) {
          int idx = r * 8 + v;
          float GI = gi[v] + acc[0 * 2 + r][v];
          float GF = gf[v] + acc[1 * 2 + r][v];
          float GG = gg[v] + acc[2 * 2 + r][v];
          float GO = go[v] + acc[3 * 2 + r][v];
          float ce = resetv ? 0.f : creg[idx];
          float cn = sigf(GF) * ce + sigf(GI) * tanhf(GG);
          creg[idx] = cn;
          hn[v] = sigf(GO) * tanhf(cn);
        }
        if (dst >= 0) {
          float* po = padded + ((size_t)n * Tt + dst) * Dd + ubase;
          *(float4*)(po)     = make_float4(hn[0], hn[1], hn[2], hn[3]);
          *(float4*)(po + 4) = make_float4(hn[4], hn[5], hn[6], hn[7]);
        }
        #pragma unroll
        for (int j = 0; j < 4; j++)
          hpack[r * 4 + j] = bndc ? 0u : pk2bf(hn[2 * j], hn[2 * j + 1]);
      }
    }
    __syncthreads();                           // all WMMA reads of h done
    if (n < Bb) {
      for (int r = 0; r < 2; r++) {
        int ubase = uw + r * 16 + half * 8;
        uint4 w = make_uint4(hpack[r * 4 + 0], hpack[r * 4 + 1],
                             hpack[r * 4 + 2], hpack[r * 4 + 3]);
        *(uint4*)&hbf[n][ubase] = w;           // ds_store_b128
      }
    }
  }
}

extern "C" void kernel_launch(void* const* d_in, const int* in_sizes, int n_in,
                              void* d_out, int out_size, void* d_ws, size_t ws_size,
                              hipStream_t stream) {
  (void)in_sizes; (void)n_in; (void)out_size; (void)ws_size;
  const float* x   = (const float*)d_in[0];
  const float* W1  = (const float*)d_in[1];
  const float* b1  = (const float*)d_in[2];
  const float* W2  = (const float*)d_in[3];
  const float* b2  = (const float*)d_in[4];
  const float* Wih = (const float*)d_in[5];
  const float* Whh = (const float*)d_in[6];
  const float* bih = (const float*)d_in[7];
  const float* bhh = (const float*)d_in[8];

  char* ws = (char*)d_ws;
  float*          gates = (float*)(ws + WS_GATES);
  unsigned char*  bnd   = (unsigned char*)(ws + WS_BND);
  int*            dest  = (int*)(ws + WS_DEST);
  int*            fb    = (int*)(ws + WS_FB);
  float*          mean  = (float*)(ws + WS_MEAN);
  unsigned short* whbf  = (unsigned short*)(ws + WS_WHBF);
  unsigned short* xbf   = (unsigned short*)(ws + WS_XBF);
  unsigned short* w1bf  = (unsigned short*)(ws + WS_W1BF);
  unsigned short* wihbf = (unsigned short*)(ws + WS_WIHBF);

  float* out    = (float*)d_out;
  float* padded = out;                                   // [B,T,D]
  float* bidx   = out + (size_t)Bb * Tt * Dd;            // [B,T]
  float* logits = bidx + (size_t)Bb * Tt;                // [B,T,2]

  // one-time bf16 packing
  k_cvt_bf16<<<32768, 256, 0, stream>>>(x,   (unsigned int*)xbf,   (Bb * Tt * Dd) / 2);
  k_cvt_bf16<<<2048,  256, 0, stream>>>(Whh, (unsigned int*)whbf,  (G4 * Dd) / 2);
  k_cvt_bf16<<<2048,  256, 0, stream>>>(Wih, (unsigned int*)wihbf, (G4 * Dd) / 2);
  k_cvt_bf16<<<256,   256, 0, stream>>>(W1,  (unsigned int*)w1bf,  (H1 * Dd) / 2);

  k_zero<<<2048, 256, 0, stream>>>((float4*)padded, (Bb * Tt * Dd) / 4);
  k_boundary_mlp<<<(Bb * Tt) / 16, 128, 0, stream>>>(xbf, w1bf, b1, W2, b2, logits, bnd);
  k_gates_gemm<<<dim3((Bb * Tt) / 128, G4 / 256), 256, 0, stream>>>(xbf, wihbf, bih, bhh, gates);
  k_scan<<<1, 32, 0, stream>>>(bnd, dest, fb, bidx);
  k_mean<<<Bb, Dd, 0, stream>>>(x, mean);
  k_fallback<<<Bb, Dd, 0, stream>>>(fb, mean, padded);
  k_seg_lstm<<<128, 512, 0, stream>>>(whbf, gates, bnd, dest, padded);
}
